// GraphAttentionLayer_21337397526655
// MI455X (gfx1250) — compile-verified, matrix operand order checked by reading, not run on hardware
//
#include <hip/hip_runtime.h>
#include <cstddef>

// ---------------------------------------------------------------------------
// GAT layer for MI455X (gfx1250, wave32, WMMA).
// Sizes fixed by the harness: B=4, N=2048, IN=128, F=64, H=4 (FH=256).
// d_out = h_prime [B,N,F] f32  ++  attention.mean(heads) [B,N,N] f32
// ---------------------------------------------------------------------------

typedef __attribute__((ext_vector_type(2)))  float  v2f;
typedef __attribute__((ext_vector_type(4)))  float  v4f;
typedef __attribute__((ext_vector_type(8)))  float  v8f;
typedef __attribute__((ext_vector_type(4)))  __bf16 v4bf;
typedef __attribute__((ext_vector_type(8)))  __bf16 v8bf;
typedef __attribute__((ext_vector_type(16))) __bf16 v16bf;
typedef __attribute__((ext_vector_type(4)))  int    v4i;

constexpr int   Bn = 4, Nn = 2048, INn = 128, Fn = 64, Hn = 4, FHn = 256;
constexpr float NEG_INF = -1.0e9f;
constexpr float SLOPE   = 0.2f;

// ---------------------------------------------------------------------------
// Kernel 1: g = h @ W  via V_WMMA_F32_16X16X4_F32 (full fp32: feeds softmax
// logits). One wave per 16x16 output tile; K-loop over IN=128 in steps of 4.
// Result stored transposed+bf16 as gT[b][h][f][q] so the big bf16 GEMM can
// fetch WMMA B-matrix registers with two contiguous b128 loads per lane.
// ---------------------------------------------------------------------------
__global__ __launch_bounds__(128) void k_gemm_g(const float* __restrict__ h,
                                                const float* __restrict__ W,
                                                __bf16* __restrict__ gT) {
  const int lane = threadIdx.x & 31;
  const int wave = threadIdx.x >> 5;
  const int tile = blockIdx.x * 4 + wave;       // 8192 tiles total
  const int b    = tile >> 11;                  // / (128 n-tiles * 16 fh-tiles)
  const int rem  = tile & 2047;
  const int n0   = (rem >> 4) << 4;             // node tile base
  const int f0   = (rem & 15) << 4;             // fh tile base
  const int col  = lane & 15;
  const int khi  = (lane >> 4) << 1;            // lanes16-31 hold K+2,K+3

  const float* hrow = h + ((size_t)b * Nn + n0 + col) * INn; // A: row = node
  const float* wcol = W + f0 + col;                          // B: col = fh

  v8f c = {};
#pragma unroll
  for (int k = 0; k < INn; k += 4) {
    const int ks = k + khi;
    v2f a = *reinterpret_cast<const v2f*>(hrow + ks);        // K, K+1
    v2f bb;
    bb.x = wcol[(size_t)ks * FHn];
    bb.y = wcol[(size_t)(ks + 1) * FHn];
    c = __builtin_amdgcn_wmma_f32_16x16x4_f32(false, a, false, bb,
                                              (short)0, c, false, false);
  }

  // C layout: vgpr v, lane l -> node = n0 + v + 8*(l/16), fh = f0 + l%16
  const int fh = f0 + col;
  const int hd = fh >> 6;
  const int f  = fh & 63;
  __bf16* out = gT + (((size_t)(b * Hn + hd) * Fn + f) * Nn) + n0 + 8 * (lane >> 4);
  v8bf vb;
#pragma unroll
  for (int v = 0; v < 8; ++v) vb[v] = (__bf16)c[v];
  *reinterpret_cast<v8bf*>(out) = vb;                        // 16B store, contiguous in q
}

// ---------------------------------------------------------------------------
// Kernel 2: per-node attention scores. s_dst = g . a1, s_src = g . a2.
// Threads cover consecutive q -> coalesced reads of gT rows.
// ---------------------------------------------------------------------------
__global__ __launch_bounds__(256) void k_scores(const __bf16* __restrict__ gT,
                                                const float* __restrict__ a1,
                                                const float* __restrict__ a2,
                                                float* __restrict__ sdst,
                                                float* __restrict__ ssrc) {
  const int bh = blockIdx.x >> 3;                      // (b*H + h)
  const int n  = ((blockIdx.x & 7) << 8) + threadIdx.x;
  const __bf16* gp = gT + (size_t)bh * Fn * Nn + n;
  float acc1 = 0.f, acc2 = 0.f;
#pragma unroll 8
  for (int f = 0; f < Fn; ++f) {
    const float g = (float)gp[(size_t)f * Nn];
    acc1 += g * a1[f];
    acc2 += g * a2[f];
  }
  sdst[bh * Nn + n] = acc1;
  ssrc[bh * Nn + n] = acc2;
}

// ---------------------------------------------------------------------------
// Kernel 3: softmax row statistics (online max / sum-of-exp) for all 4 heads
// while reading the 64MB adj matrix exactly once. One wave per row.
// ---------------------------------------------------------------------------
__global__ __launch_bounds__(256) void k_stats(const int* __restrict__ adj,
                                               const float* __restrict__ sdst,
                                               const float* __restrict__ ssrc,
                                               float* __restrict__ rmax,
                                               float* __restrict__ rsum) {
  const int lane = threadIdx.x & 31;
  const int wave = threadIdx.x >> 5;
  const int b    = blockIdx.x >> 8;                 // 256 blocks per batch
  const int p    = ((blockIdx.x & 255) << 3) + wave;
  const int* arow = adj + ((size_t)b * Nn + p) * Nn;

  float sr[Hn], m[Hn], s[Hn];
#pragma unroll
  for (int hd = 0; hd < Hn; ++hd) {
    sr[hd] = ssrc[(b * Hn + hd) * Nn + p];
    m[hd]  = -3.0e38f;
    s[hd]  = 0.f;
  }
  for (int q0 = lane * 4; q0 < Nn; q0 += 128) {
    const v4i a4 = *reinterpret_cast<const v4i*>(arow + q0);
#pragma unroll
    for (int i = 0; i < 4; ++i) {
#pragma unroll
      for (int hd = 0; hd < Hn; ++hd) {
        float v = sr[hd] + sdst[(b * Hn + hd) * Nn + q0 + i];
        v = (v >= 0.f) ? v : SLOPE * v;
        v = a4[i] ? v : NEG_INF;
        if (v > m[hd]) { s[hd] = s[hd] * __expf(m[hd] - v) + 1.f; m[hd] = v; }
        else           { s[hd] += __expf(v - m[hd]); }
      }
    }
  }
  // wave32 tree merge of (max, sum) pairs
#pragma unroll
  for (int off = 16; off > 0; off >>= 1) {
#pragma unroll
    for (int hd = 0; hd < Hn; ++hd) {
      const float mo = __shfl_xor(m[hd], off);
      const float so = __shfl_xor(s[hd], off);
      if (mo > m[hd]) { s[hd] = s[hd] * __expf(m[hd] - mo) + so; m[hd] = mo; }
      else            { s[hd] += so * __expf(mo - m[hd]); }
    }
  }
  if (lane == 0) {
#pragma unroll
    for (int hd = 0; hd < Hn; ++hd) {
      rmax[(b * Hn + hd) * Nn + p] = m[hd];
      rsum[(b * Hn + hd) * Nn + p] = s[hd];
    }
  }
}

// ---------------------------------------------------------------------------
// Kernel 4: the heavy pass. Per block: 32 p-rows of one batch.
// Per 32-wide q chunk:
//   * 256 threads build bf16 attention tiles for all 4 heads in LDS
//     (A-matrix layout) + write the head-averaged attention chunk to d_out
//   * 8 waves (2 p-subtiles x 4 f-subtiles) each run 4x
//     v_wmma_f32_16x16x32_bf16 (one per head), accumulating h_prime in f32.
// ---------------------------------------------------------------------------
__global__ __launch_bounds__(256) void k_attn(const int* __restrict__ adj,
                                              const __bf16* __restrict__ gT,
                                              const float* __restrict__ sdst,
                                              const float* __restrict__ ssrc,
                                              const float* __restrict__ rmax,
                                              const float* __restrict__ rsum,
                                              float* __restrict__ hprime,
                                              float* __restrict__ amean) {
  constexpr int ROWB = 80;                       // padded LDS row stride (bank-safe, 16B aligned)
  __shared__ __align__(16) char attls[Hn * 32 * ROWB];
  __shared__ float sdsh[Hn][32];

  const int tid  = threadIdx.x;
  const int lane = tid & 31;
  const int wave = tid >> 5;
  const int b    = blockIdx.x >> 6;              // 64 p-tiles per batch
  const int p0   = (blockIdx.x & 63) << 5;

  // ---- producer-side mapping: thread -> (p row, 4 q columns) ----
  const int pl = tid >> 3;                       // 0..31
  const int qq = (tid & 7) << 2;                 // 0,4,...,28
  const int p  = p0 + pl;
  const int* arow = adj + ((size_t)b * Nn + p) * Nn;

  float m[Hn], inv[Hn], sr[Hn];
#pragma unroll
  for (int hd = 0; hd < Hn; ++hd) {
    const int idx = (b * Hn + hd) * Nn + p;
    m[hd]   = rmax[idx];
    inv[hd] = 1.f / rsum[idx];
    sr[hd]  = ssrc[idx];
  }

  // ---- consumer-side mapping: wave -> (p subtile, f subtile) ----
  const int psub  = wave >> 2;                   // 0..1
  const int fsub  = wave & 3;                    // 0..3
  const int fcol  = (fsub << 4) + (lane & 15);
  const int base  = (lane < 16) ? 0 : 8;         // K-run base per WMMA layout
  const char* aptr0 = attls + (psub * 16 + (lane & 15)) * ROWB + base * 2;
  const __bf16* gbase[Hn];
#pragma unroll
  for (int hd = 0; hd < Hn; ++hd)
    gbase[hd] = gT + ((size_t)(b * Hn + hd) * Fn + fcol) * Nn;

  float* am = amean + ((size_t)b * Nn + p) * Nn + qq;
  v8f c = {};

  for (int q0 = 0; q0 < Nn; q0 += 32) {
    // stage this chunk's s_dst values (shared by all 32 p-rows)
    if (tid < 128)
      sdsh[tid >> 5][tid & 31] = sdst[(b * Hn + (tid >> 5)) * Nn + q0 + (tid & 31)];
    __syncthreads();

    // build bf16 attention tiles + head-mean output
    const v4i a4 = *reinterpret_cast<const v4i*>(arow + q0 + qq);
    v4f mean4 = {0.f, 0.f, 0.f, 0.f};
#pragma unroll
    for (int hd = 0; hd < Hn; ++hd) {
      v4bf av;
#pragma unroll
      for (int i = 0; i < 4; ++i) {
        float v = sr[hd] + sdsh[hd][qq + i];
        v = (v >= 0.f) ? v : SLOPE * v;
        v = a4[i] ? v : NEG_INF;
        const float at = __expf(v - m[hd]) * inv[hd];
        av[i]    = (__bf16)at;
        mean4[i] += at;
      }
      *reinterpret_cast<v4bf*>(attls + hd * (32 * ROWB) + pl * ROWB + qq * 2) = av;
    }
#pragma unroll
    for (int i = 0; i < 4; ++i) mean4[i] *= 0.25f;   // 1/H
    *reinterpret_cast<v4f*>(am + q0) = mean4;
    if (q0 + 32 < Nn) __builtin_prefetch(arow + q0 + 32 + qq, 0, 0);
    __syncthreads();

    // WMMA: h_prime[16p x 16f] += attn[16p x 32q] @ g[32q x 16f], per head
#pragma unroll
    for (int hd = 0; hd < Hn; ++hd) {
      const char* ap = aptr0 + hd * (32 * ROWB);
      const v8bf alo = *reinterpret_cast<const v8bf*>(ap);        // K=base..base+7
      const v8bf ahi = *reinterpret_cast<const v8bf*>(ap + 32);   // K=base+16..+23
      const v16bf A = __builtin_shufflevector(alo, ahi,
          0, 1, 2, 3, 4, 5, 6, 7, 8, 9, 10, 11, 12, 13, 14, 15);
      const __bf16* gp = gbase[hd] + q0 + base;
      const v8bf blo = *reinterpret_cast<const v8bf*>(gp);
      const v8bf bhi = *reinterpret_cast<const v8bf*>(gp + 16);
      const v16bf Bm = __builtin_shufflevector(blo, bhi,
          0, 1, 2, 3, 4, 5, 6, 7, 8, 9, 10, 11, 12, 13, 14, 15);
      c = __builtin_amdgcn_wmma_f32_16x16x32_bf16(false, A, false, Bm,
                                                  (short)0, c, false, false);
    }
    __syncthreads();
  }

  // h_prime writeback: C layout vgpr v, lane l -> row = v + 8*(l/16), col = l%16
  const int prow = p0 + psub * 16 + 8 * (lane >> 4);
  float* hp = hprime + ((size_t)b * Nn + prow) * Fn + (fsub << 4) + (lane & 15);
#pragma unroll
  for (int v = 0; v < 8; ++v) hp[(size_t)v * Fn] = c[v] * 0.25f;  // mean over H
}

// ---------------------------------------------------------------------------
extern "C" void kernel_launch(void* const* d_in, const int* in_sizes, int n_in,
                              void* d_out, int out_size, void* d_ws, size_t ws_size,
                              hipStream_t stream) {
  const float* h   = (const float*)d_in[0];
  const int*   adj = (const int*)d_in[1];
  const float* W   = (const float*)d_in[2];
  const float* a1  = (const float*)d_in[3];
  const float* a2  = (const float*)d_in[4];

  char* ws = (char*)d_ws;
  __bf16* gT  = reinterpret_cast<__bf16*>(ws);                    // 4 MB
  float* sdst = (float*)(ws + (size_t)Bn * Hn * Fn * Nn * 2);
  float* ssrc = sdst + Bn * Hn * Nn;
  float* rmax = ssrc + Bn * Hn * Nn;
  float* rsum = rmax + Bn * Hn * Nn;

  float* hprime = (float*)d_out;                                  // [B,N,F]
  float* amean  = hprime + (size_t)Bn * Nn * Fn;                  // [B,N,N]

  k_gemm_g<<<2048, 128, 0, stream>>>(h, W, gT);
  k_scores<<<128, 256, 0, stream>>>(gT, a1, a2, sdst, ssrc);
  k_stats <<<1024, 256, 0, stream>>>(adj, sdst, ssrc, rmax, rsum);
  k_attn  <<<256, 256, 0, stream>>>(adj, gT, sdst, ssrc, rmax, rsum, hprime, amean);
}
